// DeepFM_72189810311759
// MI455X (gfx1250) — compile-verified, matrix operand order checked by reading, not run on hardware
//
#include <hip/hip_runtime.h>

// ---------- bf16 helpers ----------
typedef __attribute__((ext_vector_type(16))) __bf16 v16bf;
typedef __attribute__((ext_vector_type(2)))  __bf16 v2bf;
typedef __attribute__((ext_vector_type(8)))  float  v8f;

// native f32 -> bf16 (clang fptrunc; backend picks packed cvt where available)
__device__ __forceinline__ unsigned short f2bf(float f) {
  __bf16 h = (__bf16)f;
  return __builtin_bit_cast(unsigned short, h);
}
__device__ __forceinline__ float bf2f(unsigned short s) {
  __bf16 h = __builtin_bit_cast(__bf16, s);
  return (float)h;
}
// packed f32x2 -> bf16x2 in one dword
__device__ __forceinline__ unsigned pkbf2(float lo, float hi) {
  v2bf r;
  r.x = (__bf16)lo;
  r.y = (__bf16)hi;
  return __builtin_bit_cast(unsigned, r);
}

union Frag { v16bf v; uint4 q[2]; };

// ---------------- constants ----------------
#define BATCH 16384
#define KDIM  512          // embedding K and MLP1 K
#define NPAD  384          // 256 dense + 16 folded-sum + 1 last + pad to 3*128

// ---------------- prep kernels ----------------
// Wm: [512][257] row-major f32 -> Wt: [384][512] bf16 (transposed + augmented)
__global__ void prep_emb_w(const float* __restrict__ W, unsigned short* __restrict__ Wt) {
  int n = blockIdx.x;        // 0..383
  int k = threadIdx.x;       // 0..511
  float v = 0.f;
  if (n < 256)       v = W[(size_t)k * 257 + n];
  else if (n < 272) { int c = n - 256;
    #pragma unroll
    for (int i = 0; i < 16; ++i) v += W[(size_t)k * 257 + i * 16 + c];
  }
  else if (n == 272) v = W[(size_t)k * 257 + 256];
  Wt[(size_t)n * KDIM + k] = f2bf(v);
}

__global__ void prep_emb_bias(const float* __restrict__ b, float* __restrict__ baug) {
  int n = threadIdx.x;       // 0..383
  float v = 0.f;
  if (n < 256)       v = b[n];
  else if (n < 272) { int c = n - 256;
    #pragma unroll
    for (int i = 0; i < 16; ++i) v += b[i * 16 + c];
  }
  else if (n == 272) v = b[256];
  baug[n] = v;
}

// W: [K][N] f32 -> Wt: [N][K] bf16.  grid = N blocks, block = K threads
__global__ void prep_transpose(const float* __restrict__ W, unsigned short* __restrict__ Wt, int N) {
  int n = blockIdx.x, k = threadIdx.x, K = blockDim.x;
  Wt[(size_t)n * K + k] = f2bf(W[(size_t)k * N + n]);
}

// ---------------- block GEMM: C[M][N] = A[M][K] * Bt[N][K]^T  (bf16 WMMA, f32 acc) ----
// Block 256 threads = 8 waves; block tile 128(M) x 128(N); wave tile 64 x 32.
// Double-buffered LDS + register prefetch: one barrier per K-step, global latency
// overlapped with the WMMA pipe.
// mode 0: embedding epilogue (bf16 all_dense cols<256, f32 sm cols 256..271, f32 last col 272)
// mode 1: relu -> bf16 out
template<bool AF32>
__global__ __launch_bounds__(256) void gemm_bf16_wmma(
    const void* __restrict__ Ap, const unsigned short* __restrict__ Bt,
    const float* __restrict__ bias,
    int M, int K, int Nlogical, int mode,
    unsigned short* __restrict__ out_bf, int out_ld, int out_col_ofs,
    float* __restrict__ sm, float* __restrict__ lastv)
{
  __shared__ unsigned short sA[2][128 * 32];
  __shared__ unsigned short sB[2][128 * 32];

  const int tid  = threadIdx.x;
  const int lane = tid & 31;
  const int wave = tid >> 5;
  const int wm   = wave >> 2;          // 0..1  -> M offset wm*64
  const int wn   = wave & 3;           // 0..3  -> N offset wn*32
  const int m_l  = lane & 15;
  const int khi  = lane >> 4;

  const int mBase = blockIdx.x * 128;
  const int nBase = blockIdx.y * 128;

  const float*          Af = (const float*)Ap;
  const unsigned short* Ab = (const unsigned short*)Ap;

  // staging geometry: 4 chunks of 4 elements per thread for each of A and B
  int sofs[4];
  const float*          aPtrF[4];
  const unsigned short* aPtrH[4];
  const unsigned short* bPtr[4];
  #pragma unroll
  for (int it = 0; it < 4; ++it) {
    int c    = tid + it * 256;
    int row  = c >> 3;
    int col4 = (c & 7) * 4;
    sofs[it] = row * 32 + col4;
    if constexpr (AF32) aPtrF[it] = Af + (size_t)(mBase + row) * K + col4;
    else                aPtrH[it] = Ab + (size_t)(mBase + row) * K + col4;
    bPtr[it] = Bt + (size_t)(nBase + row) * K + col4;
  }

  float4 aRegF[4];
  uint2  aRegH[4];
  uint2  bReg[4];

  auto loadGlobal = [&](int k0) {
    #pragma unroll
    for (int it = 0; it < 4; ++it) {
      if constexpr (AF32) aRegF[it] = *(const float4*)(aPtrF[it] + k0);
      else                aRegH[it] = *(const uint2*)(aPtrH[it] + k0);
      bReg[it] = *(const uint2*)(bPtr[it] + k0);
    }
  };
  auto storeLds = [&](int buf) {
    #pragma unroll
    for (int it = 0; it < 4; ++it) {
      if constexpr (AF32) {
        uint2 d;
        d.x = pkbf2(aRegF[it].x, aRegF[it].y);
        d.y = pkbf2(aRegF[it].z, aRegF[it].w);
        *(uint2*)&sA[buf][sofs[it]] = d;
      } else {
        *(uint2*)&sA[buf][sofs[it]] = aRegH[it];
      }
      *(uint2*)&sB[buf][sofs[it]] = bReg[it];
    }
  };

  v8f acc[4][2] = {};

  loadGlobal(0);
  const int nsteps = K >> 5;
  for (int s = 0; s < nsteps; ++s) {
    const int buf = s & 1;
    storeLds(buf);
    __syncthreads();
    if (s + 1 < nsteps) loadGlobal((s + 1) << 5);   // prefetch; wait lands next iter

    // fragments: ISA 16-bit A 16x32 layout => two contiguous 16B ds_load_b128 per lane
    Frag af[4];
    #pragma unroll
    for (int i = 0; i < 4; ++i) {
      int r = wm * 64 + i * 16 + m_l;
      af[i].q[0] = *(const uint4*)&sA[buf][r * 32 + khi * 8];
      af[i].q[1] = *(const uint4*)&sA[buf][r * 32 + 16 + khi * 8];
    }
    Frag bfr[2];
    #pragma unroll
    for (int j = 0; j < 2; ++j) {
      int r = wn * 32 + j * 16 + m_l;
      bfr[j].q[0] = *(const uint4*)&sB[buf][r * 32 + khi * 8];
      bfr[j].q[1] = *(const uint4*)&sB[buf][r * 32 + 16 + khi * 8];
    }
    #pragma unroll
    for (int i = 0; i < 4; ++i)
      #pragma unroll
      for (int j = 0; j < 2; ++j)
        acc[i][j] = __builtin_amdgcn_wmma_f32_16x16x32_bf16(
            false, af[i].v, false, bfr[j].v, (short)0, acc[i][j], false, false);
  }

  // ---- epilogue: C layout => VGPR r, lane L: M = r + 8*(L>>4), N = L&15 ----
  #pragma unroll
  for (int j = 0; j < 2; ++j) {
    int n = nBase + wn * 32 + j * 16 + m_l;
    if (n >= Nlogical) continue;
    float bv = bias ? bias[n] : 0.f;
    #pragma unroll
    for (int i = 0; i < 4; ++i) {
      #pragma unroll
      for (int r = 0; r < 8; ++r) {
        int m = mBase + wm * 64 + i * 16 + r + 8 * khi;
        float val = acc[i][j][r] + bv;
        if (mode == 0) {
          if (n < 256)       out_bf[(size_t)m * out_ld + out_col_ofs + n] = f2bf(val);
          else if (n < 272)  sm[(size_t)m * 16 + (n - 256)] = val;
          else               lastv[m] = val;                    // n == 272
        } else {
          val = val > 0.f ? val : 0.f;
          out_bf[(size_t)m * out_ld + n] = f2bf(val);
        }
      }
    }
  }
}

// ---------------- finalizer: out = h2 . W3 + b3 + last_m + last_u + dot(sm_m, sm_u) ----
__global__ void fm_final(const unsigned short* __restrict__ h2, const float* __restrict__ W3,
                         const float* __restrict__ b3,
                         const float* __restrict__ sm_m, const float* __restrict__ sm_u,
                         const float* __restrict__ last_m, const float* __restrict__ last_u,
                         float* __restrict__ out, int Bn) {
  int b = blockIdx.x * 256 + threadIdx.x;
  if (b >= Bn) return;
  const unsigned short* h = h2 + (size_t)b * 128;
  float acc = 0.f;
  #pragma unroll 4
  for (int k = 0; k < 128; ++k) acc += bf2f(h[k]) * W3[k];
  float fm = 0.f;
  const float* a = sm_m + (size_t)b * 16;
  const float* c = sm_u + (size_t)b * 16;
  #pragma unroll
  for (int k = 0; k < 16; ++k) fm += a[k] * c[k];
  out[b] = acc + b3[0] + last_m[b] + last_u[b] + fm;
}

// ---------------- launch ----------------
extern "C" void kernel_launch(void* const* d_in, const int* in_sizes, int n_in,
                              void* d_out, int out_size, void* d_ws, size_t ws_size,
                              hipStream_t stream) {
  const float* mv = (const float*)d_in[0];
  const float* uv = (const float*)d_in[1];
  const float* Wm = (const float*)d_in[2];
  const float* bm = (const float*)d_in[3];
  const float* Wu = (const float*)d_in[4];
  const float* bu = (const float*)d_in[5];
  const float* W1 = (const float*)d_in[6];
  const float* b1 = (const float*)d_in[7];
  const float* W2 = (const float*)d_in[8];
  const float* b2 = (const float*)d_in[9];
  const float* W3 = (const float*)d_in[10];
  const float* b3 = (const float*)d_in[11];
  float* out = (float*)d_out;

  // workspace carve-up (256B aligned)
  char* p = (char*)d_ws;
  size_t off = 0;
  auto alloc = [&](size_t bytes) { void* r = p + off; off = (off + bytes + 255) & ~(size_t)255; return r; };
  unsigned short* wt_m = (unsigned short*)alloc((size_t)NPAD * KDIM * 2);
  unsigned short* wt_u = (unsigned short*)alloc((size_t)NPAD * KDIM * 2);
  unsigned short* wt_1 = (unsigned short*)alloc((size_t)256 * 512 * 2);
  unsigned short* wt_2 = (unsigned short*)alloc((size_t)128 * 256 * 2);
  float* bias_m = (float*)alloc((size_t)NPAD * 4);
  float* bias_u = (float*)alloc((size_t)NPAD * 4);
  unsigned short* all_dense = (unsigned short*)alloc((size_t)BATCH * 512 * 2);
  float* sm_m   = (float*)alloc((size_t)BATCH * 16 * 4);
  float* sm_u   = (float*)alloc((size_t)BATCH * 16 * 4);
  float* last_m = (float*)alloc((size_t)BATCH * 4);
  float* last_u = (float*)alloc((size_t)BATCH * 4);
  unsigned short* h1 = (unsigned short*)alloc((size_t)BATCH * 256 * 2);
  unsigned short* h2 = (unsigned short*)alloc((size_t)BATCH * 128 * 2);
  (void)ws_size; (void)in_sizes; (void)n_in; (void)out_size;

  // weight prep (transpose + bf16 + FM augmentation)
  prep_emb_w<<<NPAD, KDIM, 0, stream>>>(Wm, wt_m);
  prep_emb_w<<<NPAD, KDIM, 0, stream>>>(Wu, wt_u);
  prep_emb_bias<<<1, NPAD, 0, stream>>>(bm, bias_m);
  prep_emb_bias<<<1, NPAD, 0, stream>>>(bu, bias_u);
  prep_transpose<<<256, 512, 0, stream>>>(W1, wt_1, 256);
  prep_transpose<<<128, 256, 0, stream>>>(W2, wt_2, 128);

  // embedding towers (augmented N=273 logical, padded to 384)
  dim3 gEmb(BATCH / 128, NPAD / 128);
  gemm_bf16_wmma<true><<<gEmb, 256, 0, stream>>>(
      mv, wt_m, bias_m, BATCH, KDIM, 273, 0, all_dense, 512, 0,   sm_m, last_m);
  gemm_bf16_wmma<true><<<gEmb, 256, 0, stream>>>(
      uv, wt_u, bias_u, BATCH, KDIM, 273, 0, all_dense, 512, 256, sm_u, last_u);

  // MLP layer 1: [B,512] -> relu -> [B,256]
  dim3 g1(BATCH / 128, 2);
  gemm_bf16_wmma<false><<<g1, 256, 0, stream>>>(
      all_dense, wt_1, b1, BATCH, 512, 256, 1, h1, 256, 0, nullptr, nullptr);

  // MLP layer 2: [B,256] -> relu -> [B,128]
  dim3 g2(BATCH / 128, 1);
  gemm_bf16_wmma<false><<<g2, 256, 0, stream>>>(
      h1, wt_2, b2, BATCH, 256, 128, 1, h2, 128, 0, nullptr, nullptr);

  // final GEMV + FM terms
  fm_final<<<BATCH / 256, 256, 0, stream>>>(h2, W3, b3, sm_m, sm_u, last_m, last_u, out, BATCH);
}
